// TransformerBlock_33621003993531
// MI455X (gfx1250) — compile-verified, matrix operand order checked by reading, not run on hardware
//
#include <hip/hip_runtime.h>

// ---------------------------------------------------------------------------
// MI455X (gfx1250) transformer block: all matmuls via v_wmma_f32_16x16x32_bf16
// ---------------------------------------------------------------------------

typedef __bf16 bf16_t;
typedef __attribute__((ext_vector_type(16))) __bf16 v16bf;
typedef __attribute__((ext_vector_type(8)))  __bf16 v8bf;
typedef __attribute__((ext_vector_type(8)))  float  v8f;

static constexpr int S_LEN   = 2048;
static constexpr int D_DIM   = 1024;
static constexpr int N_HEADS = 16;
static constexpr int HEAD_D  = 64;

// ---- fragment helpers ------------------------------------------------------
// bf16 A/B fragment (16x32 / 32x16): lane holds row (lane&15), K values at
// base (lane>>4)*8: [base, base+8) and [16+base, 16+base+8). Both halves are
// contiguous 16-byte loads from a K-contiguous row.
__device__ __forceinline__ v16bf mk_frag(v8bf lo, v8bf hi) {
    v16bf f;
#pragma unroll
    for (int i = 0; i < 8; ++i) { f[i] = lo[i]; f[i + 8] = hi[i]; }
    return f;
}

__device__ __forceinline__ v16bf load_frag(const bf16_t* p) {
    // p = row base + per-lane K base; second half is +16 elements
    return mk_frag(*(const v8bf*)p, *(const v8bf*)(p + 16));
}

__device__ __forceinline__ v8f wmma_bf16(v16bf a, v16bf b, v8f c) {
    return __builtin_amdgcn_wmma_f32_16x16x32_bf16(false, a, false, b,
                                                   (short)0, c, false, false);
}

// ---- fp32 -> bf16 convert --------------------------------------------------
__global__ void cvt_f32_bf16(const float* __restrict__ in,
                             bf16_t* __restrict__ out, int n) {
    for (int i = blockIdx.x * 256 + threadIdx.x; i < n; i += gridDim.x * 256)
        out[i] = (bf16_t)in[i];
}

// ---- LayerNorm (row-wise over 1024), fp32 in -> bf16 out -------------------
__global__ __launch_bounds__(256)
void layernorm_bf16(const float* __restrict__ x, const float* __restrict__ g,
                    const float* __restrict__ b, bf16_t* __restrict__ out) {
    const int row = blockIdx.x;
    const int tid = threadIdx.x;
    const float* xr = x + (size_t)row * D_DIM;

    float v[4];
    float s = 0.f;
#pragma unroll
    for (int i = 0; i < 4; ++i) { v[i] = xr[tid + i * 256]; s += v[i]; }
#pragma unroll
    for (int m = 1; m < 32; m <<= 1) s += __shfl_xor(s, m);

    __shared__ float red0[8], red1[8];
    if ((tid & 31) == 0) red0[tid >> 5] = s;
    __syncthreads();
    float mean = 0.f;
#pragma unroll
    for (int i = 0; i < 8; ++i) mean += red0[i];
    mean *= (1.0f / D_DIM);

    float vs = 0.f;
#pragma unroll
    for (int i = 0; i < 4; ++i) { float d = v[i] - mean; vs += d * d; }
#pragma unroll
    for (int m = 1; m < 32; m <<= 1) vs += __shfl_xor(vs, m);
    if ((tid & 31) == 0) red1[tid >> 5] = vs;
    __syncthreads();
    float var = 0.f;
#pragma unroll
    for (int i = 0; i < 8; ++i) var += red1[i];
    var *= (1.0f / D_DIM);
    const float rstd = rsqrtf(var + 1e-5f);

#pragma unroll
    for (int i = 0; i < 4; ++i) {
        int c = tid + i * 256;
        out[(size_t)row * D_DIM + c] = (bf16_t)((v[i] - mean) * rstd * g[c] + b[c]);
    }
}

// ---- head-major V transpose: out[h][d][s] = in[s][co + h*64 + d] -----------
__global__ void transpose_heads(const bf16_t* __restrict__ in, int ld, int co,
                                bf16_t* __restrict__ out, int total) {
    int idx = blockIdx.x * 256 + threadIdx.x;
    if (idx >= total) return;
    int s  = idx % S_LEN;
    int hd = idx / S_LEN;
    out[idx] = in[(size_t)s * ld + co + hd];
}

// ---- WMMA GEMM: C[M,N] = A[M,K] * W[N,K]^T + bias, opt. gelu/residual -----
// block = 256 thr (8 waves), tile 128x128; wave = 32(M) x 64(N) = 2x4 WMMA tiles
__global__ __launch_bounds__(256)
void gemm_bf16(const bf16_t* __restrict__ A, const bf16_t* __restrict__ W,
               const float* __restrict__ bias, const float* __restrict__ res,
               float* __restrict__ outF, bf16_t* __restrict__ outB,
               int N, int K, int apply_gelu) {
    const int lane = threadIdx.x & 31;
    const int wid  = threadIdx.x >> 5;
    const int wm   = wid & 3;          // 0..3  (M groups of 32)
    const int wn   = wid >> 2;         // 0..1  (N groups of 64)
    const int bm   = blockIdx.y * 128 + wm * 32;
    const int bn   = blockIdx.x * 128 + wn * 64;
    const int l16  = lane & 15;
    const int kb8  = (lane >> 4) << 3;

    v8f acc[2][4];
#pragma unroll
    for (int i = 0; i < 2; ++i)
#pragma unroll
        for (int j = 0; j < 4; ++j)
#pragma unroll
            for (int r = 0; r < 8; ++r) acc[i][j][r] = 0.f;

    for (int k = 0; k < K; k += 32) {
        v16bf af[2], bf[4];
#pragma unroll
        for (int tm = 0; tm < 2; ++tm)
            af[tm] = load_frag(A + (size_t)(bm + tm * 16 + l16) * K + k + kb8);
#pragma unroll
        for (int tn = 0; tn < 4; ++tn)
            bf[tn] = load_frag(W + (size_t)(bn + tn * 16 + l16) * K + k + kb8);
#pragma unroll
        for (int tm = 0; tm < 2; ++tm)
#pragma unroll
            for (int tn = 0; tn < 4; ++tn)
                acc[tm][tn] = wmma_bf16(af[tm], bf[tn], acc[tm][tn]);
    }

    // epilogue: C layout row = r + 8*(lane>>4), col = lane&15 within 16x16 tile
    const int hl = lane >> 4;
#pragma unroll
    for (int tm = 0; tm < 2; ++tm)
#pragma unroll
        for (int tn = 0; tn < 4; ++tn)
#pragma unroll
            for (int r = 0; r < 8; ++r) {
                int rg = bm + tm * 16 + r + 8 * hl;
                int cg = bn + tn * 16 + l16;
                float v = acc[tm][tn][r] + bias[cg];
                if (apply_gelu)
                    v = 0.5f * v * (1.0f + erff(v * 0.70710678118654752f));
                if (res) v += res[(size_t)rg * N + cg];
                if (outF) outF[(size_t)rg * N + cg] = v;
                else      outB[(size_t)rg * N + cg] = (bf16_t)v;
            }
}

// ---- flash-style causal attention (one head per blockIdx.y) ----------------
// block = 256 thr (8 waves), 128 query rows/block, 16 rows/wave, 32-key steps.
// Q,K are bf16 [S, ld] with head column offset h*64 applied via base pointer;
// VT is bf16 [H][64][S]; out is bf16 [S, D].
__global__ __launch_bounds__(256)
void flash_attn(const bf16_t* __restrict__ Q, int ldq,
                const bf16_t* __restrict__ Kp, int ldk,
                const bf16_t* __restrict__ VT,
                bf16_t* __restrict__ O, float scale) {
    const int h   = blockIdx.y;
    const int qb0 = blockIdx.x * 128;
    const int tid = threadIdx.x, lane = tid & 31, wid = tid >> 5;
    const int hl  = lane >> 4, l16 = lane & 15;
    const int kb8 = hl << 3;

    __shared__ __align__(16) bf16_t sK[32 * 64];        // K rows   [32][64]
    __shared__ __align__(16) bf16_t sV[64 * 32];        // V^T rows [64][32]
    __shared__ __align__(16) bf16_t sP[8 * 16 * 32];    // per-wave P [16][32]

    const bf16_t* vt = VT + (size_t)h * HEAD_D * S_LEN;

    // Q fragments for this wave's 16 rows (K = 64 -> two fragments), loop-invariant
    const int qrow = qb0 + wid * 16 + l16;
    const bf16_t* qp = Q + (size_t)qrow * ldq + h * HEAD_D + kb8;
    const v16bf qf0 = mk_frag(*(const v8bf*)qp,        *(const v8bf*)(qp + 16));
    const v16bf qf1 = mk_frag(*(const v8bf*)(qp + 32), *(const v8bf*)(qp + 48));

    v8f oacc[4];
#pragma unroll
    for (int t = 0; t < 4; ++t)
#pragma unroll
        for (int r = 0; r < 8; ++r) oacc[t][r] = 0.f;
    float mrow[8], lrow[8];
#pragma unroll
    for (int r = 0; r < 8; ++r) { mrow[r] = -1e30f; lrow[r] = 0.f; }

    const int kend = qb0 + 128;   // uniform over the block (causal upper bound)
    for (int kb = 0; kb < kend; kb += 32) {
        __syncthreads();   // previous iteration's LDS reads complete
        {   // cooperative stage of K block and V^T block
            int r = tid >> 3, c = (tid & 7) << 3;          // 32 rows x 8 chunks
            *(v8bf*)&sK[r * 64 + c] =
                *(const v8bf*)(Kp + (size_t)(kb + r) * ldk + h * HEAD_D + c);
            int d = tid >> 2, cc = (tid & 3) << 3;         // 64 rows x 4 chunks
            *(v8bf*)&sV[d * 32 + cc] =
                *(const v8bf*)(vt + (size_t)d * S_LEN + kb + cc);
        }
        __syncthreads();

        // scores: two 16x16 tiles, K-depth 64 via two WMMAs each
        v8f sc[2];
#pragma unroll
        for (int t = 0; t < 2; ++t) {
            v8f a;
#pragma unroll
            for (int r = 0; r < 8; ++r) a[r] = 0.f;
            const bf16_t* kp = &sK[(t * 16 + l16) * 64 + kb8];
            a = wmma_bf16(qf0, mk_frag(*(const v8bf*)kp,        *(const v8bf*)(kp + 16)), a);
            a = wmma_bf16(qf1, mk_frag(*(const v8bf*)(kp + 32), *(const v8bf*)(kp + 48)), a);
            sc[t] = a;
        }

        // scale + causal mask + online softmax (fp32)
        float mnew[8];
#pragma unroll
        for (int r = 0; r < 8; ++r) {
            const int rg = qb0 + wid * 16 + r + 8 * hl;
            float s0 = sc[0][r] * scale; if (kb + l16      > rg) s0 = -1e30f;
            float s1 = sc[1][r] * scale; if (kb + 16 + l16 > rg) s1 = -1e30f;
            sc[0][r] = s0; sc[1][r] = s1;
            float m = fmaxf(s0, s1);
            m = fmaxf(m, __shfl_xor(m, 1));
            m = fmaxf(m, __shfl_xor(m, 2));
            m = fmaxf(m, __shfl_xor(m, 4));
            m = fmaxf(m, __shfl_xor(m, 8));
            mnew[r] = fmaxf(mrow[r], m);
        }
#pragma unroll
        for (int r = 0; r < 8; ++r) {
            const float f = __expf(mrow[r] - mnew[r]);
            lrow[r] *= f;
#pragma unroll
            for (int t = 0; t < 4; ++t) oacc[t][r] *= f;
            const float p0 = __expf(sc[0][r] - mnew[r]);
            const float p1 = __expf(sc[1][r] - mnew[r]);
            float ps = p0 + p1;
            ps += __shfl_xor(ps, 1);
            ps += __shfl_xor(ps, 2);
            ps += __shfl_xor(ps, 4);
            ps += __shfl_xor(ps, 8);
            lrow[r] += ps;
            mrow[r]  = mnew[r];
            // scatter P into per-wave A-operand staging buffer (C-layout position)
            sP[wid * 512 + (r + 8 * hl) * 32 + l16]      = (bf16_t)p0;
            sP[wid * 512 + (r + 8 * hl) * 32 + 16 + l16] = (bf16_t)p1;
        }
        __syncthreads();

        // PV: A = P (16x32), B = V^T tiles (32x16 each), four output tiles
        const bf16_t* pp = &sP[wid * 512 + l16 * 32 + kb8];
        const v16bf pf = mk_frag(*(const v8bf*)pp, *(const v8bf*)(pp + 16));
#pragma unroll
        for (int t = 0; t < 4; ++t) {
            const bf16_t* vp = &sV[(t * 16 + l16) * 32 + kb8];
            oacc[t] = wmma_bf16(pf, mk_frag(*(const v8bf*)vp, *(const v8bf*)(vp + 16)),
                                oacc[t]);
        }
    }

    // normalize and store ctx (bf16 [S, D])
#pragma unroll
    for (int t = 0; t < 4; ++t)
#pragma unroll
        for (int r = 0; r < 8; ++r) {
            const int rg = qb0 + wid * 16 + r + 8 * hl;
            O[(size_t)rg * D_DIM + h * HEAD_D + t * 16 + l16] =
                (bf16_t)(oacc[t][r] / lrow[r]);
        }
}

// ---------------------------------------------------------------------------
extern "C" void kernel_launch(void* const* d_in, const int* in_sizes, int n_in,
                              void* d_out, int out_size, void* d_ws, size_t ws_size,
                              hipStream_t stream) {
    const float* x     = (const float*)d_in[0];
    const float* ln1_g = (const float*)d_in[1];
    const float* ln1_b = (const float*)d_in[2];
    const float* w_qkv = (const float*)d_in[3];
    const float* b_qkv = (const float*)d_in[4];
    const float* w_ao  = (const float*)d_in[5];
    const float* b_ao  = (const float*)d_in[6];
    const float* ln2_g = (const float*)d_in[7];
    const float* ln2_b = (const float*)d_in[8];
    const float* w_q   = (const float*)d_in[9];
    const float* b_q   = (const float*)d_in[10];
    const float* w_fo  = (const float*)d_in[11];
    const float* b_fo  = (const float*)d_in[12];
    float* out = (float*)d_out;

    const int S = S_LEN, D = D_DIM, H = N_HEADS;
    char* ws = (char*)d_ws;
    size_t off = 0;
    auto alloc = [&](size_t bytes) -> void* {
        void* p = ws + off;
        off += (bytes + 255) & ~(size_t)255;
        return p;
    };
    bf16_t* xn1    = (bf16_t*)alloc((size_t)S * D * 2);
    bf16_t* qkv    = (bf16_t*)alloc((size_t)S * 3 * D * 2);
    bf16_t* vt1    = (bf16_t*)alloc((size_t)D * S * 2);
    bf16_t* ctx1   = (bf16_t*)alloc((size_t)S * D * 2);
    float*  hbuf   = (float*) alloc((size_t)S * D * 4);
    bf16_t* ln2n   = (bf16_t*)alloc((size_t)S * D * 2);
    bf16_t* vt2    = (bf16_t*)alloc((size_t)D * S * 2);
    bf16_t* q2     = (bf16_t*)alloc((size_t)S * D * 2);
    bf16_t* ctx2   = (bf16_t*)alloc((size_t)S * D * 2);
    bf16_t* wqkv_b = (bf16_t*)alloc((size_t)3 * D * D * 2);
    bf16_t* wao_b  = (bf16_t*)alloc((size_t)D * D * 2);
    bf16_t* wq_b   = (bf16_t*)alloc((size_t)D * D * 2);
    bf16_t* wfo_b  = (bf16_t*)alloc((size_t)D * D * 2);

    const float scale = 0.125f;   // 1/sqrt(64)

    // weights -> bf16
    cvt_f32_bf16<<<2048, 256, 0, stream>>>(w_qkv, wqkv_b, 3 * D * D);
    cvt_f32_bf16<<<2048, 256, 0, stream>>>(w_ao,  wao_b,  D * D);
    cvt_f32_bf16<<<2048, 256, 0, stream>>>(w_q,   wq_b,   D * D);
    cvt_f32_bf16<<<2048, 256, 0, stream>>>(w_fo,  wfo_b,  D * D);

    // h = x + attn(LN1(x))
    layernorm_bf16<<<S, 256, 0, stream>>>(x, ln1_g, ln1_b, xn1);
    gemm_bf16<<<dim3(3 * D / 128, S / 128), 256, 0, stream>>>(
        xn1, wqkv_b, b_qkv, nullptr, nullptr, qkv, 3 * D, D, 0);
    transpose_heads<<<(D * S) / 256, 256, 0, stream>>>(qkv, 3 * D, 2 * D, vt1, D * S);
    flash_attn<<<dim3(S / 128, H), 256, 0, stream>>>(
        qkv, 3 * D, qkv + D, 3 * D, vt1, ctx1, scale);
    gemm_bf16<<<dim3(D / 128, S / 128), 256, 0, stream>>>(
        ctx1, wao_b, b_ao, x, hbuf, nullptr, D, D, 0);

    // out = h + gelu(ffn_q_attn(LN2(h)))
    layernorm_bf16<<<S, 256, 0, stream>>>(hbuf, ln2_g, ln2_b, ln2n);
    transpose_heads<<<(D * S) / 256, 256, 0, stream>>>(ln2n, D, 0, vt2, D * S);
    gemm_bf16<<<dim3(D / 128, S / 128), 256, 0, stream>>>(
        ln2n, wq_b, b_q, nullptr, nullptr, q2, D, D, 0);
    flash_attn<<<dim3(S / 128, H), 256, 0, stream>>>(
        q2, D, q2, D, vt2, ctx2, scale);
    gemm_bf16<<<dim3(D / 128, S / 128), 256, 0, stream>>>(
        ctx2, wfo_b, b_fo, hbuf, out, nullptr, D, D, 1);
}
